// MultiHeadAttention_89953795047643
// MI455X (gfx1250) — compile-verified, matrix operand order checked by reading, not run on hardware
//
#include <hip/hip_runtime.h>
#include <hip/hip_bf16.h>
#include <stdint.h>

// ---------------------------------------------------------------------------
// MHA forward for MI455X (gfx1250): bf16 WMMA everywhere, fp32 accumulate.
//   B=2, S=2048, D=1024, H=16, hd=64
// Round 2: + async global->LDS copies for K tiles (ASYNCcnt path),
//          + global_prefetch_b8 for next GEMM K-step tiles.
// ---------------------------------------------------------------------------

#define D_MODEL  1024
#define NHEADS   16
#define HEAD_DIM 64
#define BATCH    2
#define SEQ      2048
#define MTOT     (BATCH * SEQ)      // 4096 rows

typedef __attribute__((ext_vector_type(8)))  float  v8f;
typedef __attribute__((ext_vector_type(16))) __bf16 v16bf;

union FragBF { v16bf v; unsigned u[8]; };

// K-pair base offset for 16-bit A/B WMMA fragments (ISA 7.12.2):
// VGPR v holds elements k, k+1 with k = 16*(v>=4) + 8*half + 2*(v&3)
__device__ __forceinline__ int kpair_base(int v, int half) {
  return ((v & 4) << 2) + (half << 3) + ((v & 3) << 1);
}

// Load a 16x32 bf16 A (or 32x16 B) fragment from an LDS tile.
// `rowp` must already point at the lane's row (row0 + (lane&15)) with given stride.
__device__ __forceinline__ v16bf load_frag(const __bf16* rowp, int k0, int half) {
  FragBF f;
#pragma unroll
  for (int v = 0; v < 8; ++v)
    f.u[v] = *(const unsigned*)(rowp + k0 + kpair_base(v, half));
  return f.v;
}

// Async 16-byte global -> LDS copy (gfx1250 GLOBAL_LOAD_ASYNC_TO_LDS_B128).
// ldsoff: LDS byte offset (low 32 bits of flat shared address);
// gaddr : 64-bit global address.
__device__ __forceinline__ void async_copy_b128(unsigned ldsoff, const void* gptr) {
  unsigned long long gaddr = (unsigned long long)gptr;
  asm volatile("global_load_async_to_lds_b128 %0, %1, off"
               :: "v"(ldsoff), "v"(gaddr)
               : "memory");
}

__device__ __forceinline__ void wait_asynccnt0() {
  asm volatile("s_wait_asynccnt 0x0" ::: "memory");
}

// ---------------------------------------------------------------------------
// Tiled WMMA GEMM:  C[M,N] = A[M,K] @ W[N,K]^T + bias
//   block tile 128x128, BK=32, 256 threads (8 waves), wave tile 32x64
// MODE 0: QKV projection epilogue -> scatter bf16 into q/k/v [B,H,S,hd]
//         (q pre-scaled by 1/sqrt(hd))
// MODE 1: plain fp32 output + bias
// ---------------------------------------------------------------------------
template <int MODE, bool A_F32>
__global__ __launch_bounds__(256)
void gemm_wmma_bf16(const void* __restrict__ Aptr,
                    const float* __restrict__ W,
                    const float* __restrict__ bias,
                    void* __restrict__ out0,
                    void* __restrict__ out1,
                    void* __restrict__ out2,
                    int Ntot, int Ktot)
{
  __shared__ __bf16 As[128][34];   // row stride 68 B (4B aligned pairs)
  __shared__ __bf16 Bs[128][34];   // Bs[n][k] = W[n0+n][k0+k]

  const int tid  = threadIdx.x;
  const int lane = tid & 31;
  const int wid  = tid >> 5;
  const int half = lane >> 4;
  const int m0 = blockIdx.y * 128;
  const int n0 = blockIdx.x * 128;
  const int wm = (wid >> 1) * 32;       // wave M offset inside tile
  const int wn = (wid & 1) * 64;        // wave N offset inside tile

  v8f acc[2][4];
#pragma unroll
  for (int i = 0; i < 2; ++i)
#pragma unroll
    for (int j = 0; j < 4; ++j)
#pragma unroll
      for (int e = 0; e < 8; ++e) acc[i][j][e] = 0.0f;

  const int lrow = tid >> 3;            // 0..31
  const int lcol = (tid & 7) * 4;       // 0..28

  for (int k0 = 0; k0 < Ktot; k0 += 32) {
    // ---- stage tiles (fp32 -> bf16 conversion happens here, for free) ----
#pragma unroll
    for (int p = 0; p < 4; ++p) {
      const int r = lrow + p * 32;
      if constexpr (A_F32) {
        const float4 fa = *(const float4*)((const float*)Aptr +
                           (size_t)(m0 + r) * Ktot + k0 + lcol);
        As[r][lcol + 0] = (__bf16)fa.x; As[r][lcol + 1] = (__bf16)fa.y;
        As[r][lcol + 2] = (__bf16)fa.z; As[r][lcol + 3] = (__bf16)fa.w;
      } else {
        const __bf16* ap = (const __bf16*)Aptr + (size_t)(m0 + r) * Ktot + k0 + lcol;
        As[r][lcol + 0] = ap[0]; As[r][lcol + 1] = ap[1];
        As[r][lcol + 2] = ap[2]; As[r][lcol + 3] = ap[3];
      }
      const float4 fb = *(const float4*)(W + (size_t)(n0 + r) * Ktot + k0 + lcol);
      Bs[r][lcol + 0] = (__bf16)fb.x; Bs[r][lcol + 1] = (__bf16)fb.y;
      Bs[r][lcol + 2] = (__bf16)fb.z; Bs[r][lcol + 3] = (__bf16)fb.w;
    }
    // ---- prefetch next K-step tiles (global_prefetch_b8) ----
    if (k0 + 32 < Ktot) {
      if constexpr (A_F32) {
        __builtin_prefetch((const float*)Aptr + (size_t)(m0 + lrow) * Ktot + k0 + 32 + lcol, 0, 3);
      } else {
        __builtin_prefetch((const __bf16*)Aptr + (size_t)(m0 + lrow) * Ktot + k0 + 32 + lcol, 0, 3);
      }
      __builtin_prefetch(W + (size_t)(n0 + lrow) * Ktot + k0 + 32 + lcol, 0, 3);
    }
    __syncthreads();

    // ---- fragments + 8 WMMAs per wave per K-step ----
    v16bf afrag[2], bfrag[4];
#pragma unroll
    for (int i = 0; i < 2; ++i)
      afrag[i] = load_frag(&As[wm + i * 16 + (lane & 15)][0], 0, half);
#pragma unroll
    for (int j = 0; j < 4; ++j)
      bfrag[j] = load_frag(&Bs[wn + j * 16 + (lane & 15)][0], 0, half);
#pragma unroll
    for (int i = 0; i < 2; ++i)
#pragma unroll
      for (int j = 0; j < 4; ++j)
        acc[i][j] = __builtin_amdgcn_wmma_f32_16x16x32_bf16(
            false, afrag[i], false, bfrag[j], (short)0, acc[i][j], false, false);
    __syncthreads();
  }

  // ---- epilogue ----
#pragma unroll
  for (int i = 0; i < 2; ++i) {
#pragma unroll
    for (int j = 0; j < 4; ++j) {
#pragma unroll
      for (int r = 0; r < 8; ++r) {
        const int row = m0 + wm + i * 16 + r + half * 8;
        const int col = n0 + wn + j * 16 + (lane & 15);
        float vv = acc[i][j][r] + bias[col];
        if constexpr (MODE == 0) {
          const int which = col >> 10;          // 0=q 1=k 2=v
          const int d     = col & 1023;
          const int h     = d >> 6;
          const int dd    = d & 63;
          const int bb    = row >> 11;          // S = 2048
          const int s     = row & 2047;
          const size_t idx = (((size_t)bb * NHEADS + h) * SEQ + s) * HEAD_DIM + dd;
          if (which == 0)      ((__bf16*)out0)[idx] = (__bf16)(vv * 0.125f); // 1/sqrt(64)
          else if (which == 1) ((__bf16*)out1)[idx] = (__bf16)vv;
          else                 ((__bf16*)out2)[idx] = (__bf16)vv;
        } else {
          ((float*)out0)[(size_t)row * Ntot + col] = vv;
        }
      }
    }
  }
}

// ---------------------------------------------------------------------------
// Flash attention (causal), bf16 WMMA, fp32 softmax state.
//   grid: (S/64, B*H), block: 128 threads = 4 waves
//   wave -> one 16-row q tile; block shares 64-key K/V tiles via LDS.
//   K tiles staged with async global->LDS B128 copies (ASYNCcnt).
// ---------------------------------------------------------------------------
__global__ __launch_bounds__(128)
void flash_attn_wmma(const __bf16* __restrict__ Qb,
                     const __bf16* __restrict__ Kb,
                     const __bf16* __restrict__ Vb,
                     __bf16* __restrict__ ctx)
{
  __shared__ __bf16 Ks[64][72];      // [key][d]   row 144 B (16B-aligned chunks)
  __shared__ __bf16 Vt[64][66];      // [d][key]   transposed for PV B-fragments
  __shared__ __bf16 Ps[4][16][66];   // per-wave P staging (C-layout -> A-layout)

  const int tid  = threadIdx.x;
  const int lane = tid & 31;
  const int wid  = tid >> 5;
  const int half = lane >> 4;
  const int qb0  = blockIdx.x * 64;
  const int bh   = blockIdx.y;
  const int b    = bh >> 4;
  const int h    = bh & 15;

  const __bf16* Q = Qb + (size_t)bh * SEQ * HEAD_DIM;
  const __bf16* K = Kb + (size_t)bh * SEQ * HEAD_DIM;
  const __bf16* V = Vb + (size_t)bh * SEQ * HEAD_DIM;
  const int qrow0 = qb0 + wid * 16;

  // Q fragments (16 rows x 64 d  ->  two 16x32 A-frags), scale already folded in
  v16bf qfrag[2];
  {
    const __bf16* qp = Q + (size_t)(qrow0 + (lane & 15)) * HEAD_DIM;
    qfrag[0] = load_frag(qp, 0,  half);
    qfrag[1] = load_frag(qp, 32, half);
  }

  v8f o[4];
#pragma unroll
  for (int t = 0; t < 4; ++t)
#pragma unroll
    for (int e = 0; e < 8; ++e) o[t][e] = 0.0f;
  float mrow[8], lrow[8];
#pragma unroll
  for (int r = 0; r < 8; ++r) { mrow[r] = -1e30f; lrow[r] = 0.0f; }

  for (int j0 = 0; j0 < qb0 + 64; j0 += 64) {
    // ---- K tile: async global->LDS copies (pure copy, no conversion) ----
    {
      const int r  = tid >> 1;
      const int c0 = (tid & 1) * 32;
      const __bf16* ksrc = K + (size_t)(j0 + r) * HEAD_DIM + c0;
#pragma unroll
      for (int q4 = 0; q4 < 4; ++q4) {
        const unsigned ldsoff = (unsigned)(size_t)&Ks[r][c0 + q4 * 8];
        async_copy_b128(ldsoff, ksrc + q4 * 8);
      }
      // ---- V tile: manual load + transpose into LDS ----
      const __bf16* vsrc = V + (size_t)(j0 + r) * HEAD_DIM + c0;
#pragma unroll
      for (int c = 0; c < 32; ++c) Vt[c0 + c][r] = vsrc[c];
    }
    wait_asynccnt0();
    __syncthreads();

    if (j0 <= qrow0 + 15) {   // wave has unmasked keys in this block
      // ---- S = Q @ K^T : 16x64 scores = 4 n-tiles x 2 k-steps ----
      v8f sacc[4];
#pragma unroll
      for (int t = 0; t < 4; ++t)
#pragma unroll
        for (int e = 0; e < 8; ++e) sacc[t][e] = 0.0f;
#pragma unroll
      for (int t = 0; t < 4; ++t) {
        const __bf16* krow = &Ks[t * 16 + (lane & 15)][0];
#pragma unroll
        for (int c = 0; c < 2; ++c) {
          v16bf kf = load_frag(krow, c * 32, half);
          sacc[t] = __builtin_amdgcn_wmma_f32_16x16x32_bf16(
              false, qfrag[c], false, kf, (short)0, sacc[t], false, false);
        }
      }
      // ---- causal mask (only the diagonal block needs it) ----
      if (j0 + 64 > qrow0) {
#pragma unroll
        for (int t = 0; t < 4; ++t)
#pragma unroll
          for (int r = 0; r < 8; ++r) {
            const int rowg = qrow0 + r + half * 8;
            const int colg = j0 + t * 16 + (lane & 15);
            if (colg > rowg) sacc[t][r] = -1e30f;
          }
      }
      // ---- online softmax: row stats live in 16-lane halves ----
      float alpha[8];
#pragma unroll
      for (int r = 0; r < 8; ++r) {
        float mx = sacc[0][r];
        mx = fmaxf(mx, sacc[1][r]); mx = fmaxf(mx, sacc[2][r]); mx = fmaxf(mx, sacc[3][r]);
        mx = fmaxf(mx, __shfl_xor(mx, 1, 16));
        mx = fmaxf(mx, __shfl_xor(mx, 2, 16));
        mx = fmaxf(mx, __shfl_xor(mx, 4, 16));
        mx = fmaxf(mx, __shfl_xor(mx, 8, 16));
        const float mn = fmaxf(mrow[r], mx);
        alpha[r] = __expf(mrow[r] - mn);
        mrow[r]  = mn;
      }
      float rs[8];
#pragma unroll
      for (int r = 0; r < 8; ++r) rs[r] = 0.0f;
#pragma unroll
      for (int t = 0; t < 4; ++t)
#pragma unroll
        for (int r = 0; r < 8; ++r) {
          const float p = __expf(sacc[t][r] - mrow[r]);
          rs[r] += p;
          Ps[wid][r + half * 8][t * 16 + (lane & 15)] = (__bf16)p;
        }
#pragma unroll
      for (int r = 0; r < 8; ++r) {
        float s = rs[r];
        s += __shfl_xor(s, 1, 16);
        s += __shfl_xor(s, 2, 16);
        s += __shfl_xor(s, 4, 16);
        s += __shfl_xor(s, 8, 16);
        lrow[r] = lrow[r] * alpha[r] + s;
      }
#pragma unroll
      for (int t = 0; t < 4; ++t)
#pragma unroll
        for (int r = 0; r < 8; ++r) o[t][r] *= alpha[r];

      // ---- O += P @ V : P via LDS (C-layout -> A-frag), V from transposed tile
      v16bf pfrag[2];
      {
        const __bf16* pp = &Ps[wid][lane & 15][0];
        pfrag[0] = load_frag(pp, 0,  half);
        pfrag[1] = load_frag(pp, 32, half);
      }
#pragma unroll
      for (int t2 = 0; t2 < 4; ++t2) {
        const __bf16* vrow = &Vt[t2 * 16 + (lane & 15)][0];
#pragma unroll
        for (int c = 0; c < 2; ++c) {
          v16bf vf = load_frag(vrow, c * 32, half);
          o[t2] = __builtin_amdgcn_wmma_f32_16x16x32_bf16(
              false, pfrag[c], false, vf, (short)0, o[t2], false, false);
        }
      }
    }
    __syncthreads();
  }

  // ---- normalize + write ctx as [B, S, D] bf16 ----
#pragma unroll
  for (int t2 = 0; t2 < 4; ++t2)
#pragma unroll
    for (int r = 0; r < 8; ++r) {
      const int rowg = qrow0 + r + half * 8;
      const int dcol = h * HEAD_DIM + t2 * 16 + (lane & 15);
      const float val = o[t2][r] / lrow[r];
      ctx[((size_t)b * SEQ + rowg) * D_MODEL + dcol] = (__bf16)val;
    }
}

// ---------------------------------------------------------------------------
extern "C" void kernel_launch(void* const* d_in, const int* in_sizes, int n_in,
                              void* d_out, int out_size, void* d_ws, size_t ws_size,
                              hipStream_t stream) {
  (void)in_sizes; (void)n_in; (void)out_size; (void)ws_size;
  const float* x     = (const float*)d_in[0];   // [B,S,D]
  const float* qkv_w = (const float*)d_in[1];   // [3D,D]
  const float* qkv_b = (const float*)d_in[2];   // [3D]
  const float* out_w = (const float*)d_in[3];   // [D,D]
  const float* out_b = (const float*)d_in[4];   // [D]
  float* out = (float*)d_out;                   // [B,S,D]

  const size_t per = (size_t)BATCH * NHEADS * SEQ * HEAD_DIM;   // 4,194,304
  __bf16* qbuf = (__bf16*)d_ws;
  __bf16* kbuf = qbuf + per;
  __bf16* vbuf = kbuf + per;
  __bf16* ctxb = vbuf + per;                    // [B,S,D] bf16

  // 1) fused QKV projection: [4096,1024] x [1024,3072] -> scatter q/k/v
  gemm_wmma_bf16<0, true><<<dim3(3 * D_MODEL / 128, MTOT / 128), 256, 0, stream>>>(
      x, qkv_w, qkv_b, qbuf, kbuf, vbuf, 3 * D_MODEL, D_MODEL);

  // 2) causal flash attention
  flash_attn_wmma<<<dim3(SEQ / 64, BATCH * NHEADS), 128, 0, stream>>>(
      qbuf, kbuf, vbuf, ctxb);

  // 3) output projection: [4096,1024] x [1024,1024] -> fp32 out + bias
  gemm_wmma_bf16<1, false><<<dim3(D_MODEL / 128, MTOT / 128), 256, 0, stream>>>(
      ctxb, out_w, out_b, out, nullptr, nullptr, D_MODEL, D_MODEL);
}
